// TransitionDownBlock_26121991094963
// MI455X (gfx1250) — matrix operand order repaired and checked by reading, not compiled, not run on hardware
//
#include <hip/hip_runtime.h>
#include <hip/hip_bf16.h>

typedef _Float16 f16;
typedef __attribute__((ext_vector_type(16))) _Float16 v16h;
typedef __attribute__((ext_vector_type(8)))  float    v8f;

constexpr int NB    = 8;
constexpr int NPTS  = 8192;
constexpr int DFEAT = 61;
constexpr int NS    = 2048;
constexpr int NK    = 16;
constexpr int CIN   = 64;
constexpr int COUT  = 128;
#define BN_EPS 1e-5f

#define WMMA(A, Bv, C) __builtin_amdgcn_wmma_f32_16x16x32_f16(false, (A), false, (Bv), (short)0, (C), false, false)

// ---------------- fragment helpers (contiguous k = 16*half + s order, A and B alike) ----------------

// A fragment 16x32 f16 from row-major [16][stride] LDS tile.
// lanes 0-15 hold K=0..15 (2 per VGPR), lanes 16-31 hold K=16..31
__device__ __forceinline__ v16h load_a_lds(const f16* src, int stride, int lane) {
  int half = lane >> 4, m = lane & 15;
  const f16* row = src + m * stride;
  v16h a;
#pragma unroll
  for (int v = 0; v < 8; ++v) {
    int k = 16 * half + 2 * v;
    a[2 * v]     = row[k];
    a[2 * v + 1] = row[k + 1];
  }
  return a;
}

// B fragment 32x16 f16 from row-major [K][stride] LDS tile.
// lanes 0-15 hold K=0..15 (2 per VGPR), lanes 16-31 hold K=16..31; N = lane&15
__device__ __forceinline__ v16h load_b_lds(const f16* src, int stride, int lane) {
  int half = lane >> 4, n = lane & 15;
  v16h b;
#pragma unroll
  for (int v = 0; v < 8; ++v) {
    int k = 16 * half + 2 * v;
    b[2 * v]     = src[k * stride + n];
    b[2 * v + 1] = src[(k + 1) * stride + n];
  }
  return b;
}

// B = srcT : source stored row-major [16][stride] (N rows, K cols) -> B[k][n]=src[n][k]
__device__ __forceinline__ v16h load_bT_lds(const f16* src, int stride, int lane) {
  int half = lane >> 4, n = lane & 15;
  const f16* row = src + n * stride;
  v16h b;
#pragma unroll
  for (int v = 0; v < 8; ++v) {
    int k = 16 * half + 2 * v;
    b[2 * v]     = row[k];
    b[2 * v + 1] = row[k + 1];
  }
  return b;
}

// D (f32 16x16) -> row-major f16 LDS tile. VGPR v holds row M=v+8*half, col N=lane&15.
__device__ __forceinline__ void store_d_f16(f16* dst, int stride, int lane, v8f d) {
  int half = lane >> 4, n = lane & 15;
#pragma unroll
  for (int v = 0; v < 8; ++v) dst[(v + 8 * half) * stride + n] = (f16)d[v];
}

// ---------------- kernel 1: farthest point sampling ----------------
// One block per batch; each of 1024 threads owns 8 points in registers.
__global__ void __launch_bounds__(1024) fps_kernel(const float* __restrict__ xyz,
                                                   int* __restrict__ fps_idx) {
  const int b = blockIdx.x;
  const int t = threadIdx.x;
  const float* base = xyz + (size_t)b * NPTS * 3;

  float px[8], py[8], pz[8], dd[8];
  const int i0 = t * 8;
#pragma unroll
  for (int j = 0; j < 8; ++j) {
    px[j] = base[(i0 + j) * 3 + 0];
    py[j] = base[(i0 + j) * 3 + 1];
    pz[j] = base[(i0 + j) * 3 + 2];
    dd[j] = 1e10f;
  }

  __shared__ float cen[3];
  __shared__ int   sfar;
  __shared__ float wv[32];
  __shared__ int   wi[32];

  int far = 0;
  for (int s = 0; s < NS; ++s) {
    if (t == 0) fps_idx[b * NS + s] = far;
    if (t == (far >> 3)) {
      int e = far & 7;
      float cx = 0.f, cy = 0.f, cz = 0.f;
#pragma unroll
      for (int j = 0; j < 8; ++j)
        if (e == j) { cx = px[j]; cy = py[j]; cz = pz[j]; }
      cen[0] = cx; cen[1] = cy; cen[2] = cz;
    }
    __syncthreads();
    const float cx = cen[0], cy = cen[1], cz = cen[2];

    float bv = -1.0f;
    int   bi = 0;
#pragma unroll
    for (int j = 0; j < 8; ++j) {
      float dx = px[j] - cx, dy = py[j] - cy, dz = pz[j] - cz;
      float d  = dx * dx + dy * dy + dz * dz;
      float nd = fminf(dd[j], d);
      dd[j] = nd;
      if (nd > bv) { bv = nd; bi = i0 + j; }
    }
    // wave32 argmax (lowest index wins ties)
#pragma unroll
    for (int m = 1; m < 32; m <<= 1) {
      float ov = __shfl_xor(bv, m, 32);
      int   oi = __shfl_xor(bi, m, 32);
      if (ov > bv || (ov == bv && oi < bi)) { bv = ov; bi = oi; }
    }
    const int w = t >> 5, l = t & 31;
    if (l == 0) { wv[w] = bv; wi[w] = bi; }
    __syncthreads();
    if (w == 0) {
      bv = wv[l]; bi = wi[l];
#pragma unroll
      for (int m = 1; m < 32; m <<= 1) {
        float ov = __shfl_xor(bv, m, 32);
        int   oi = __shfl_xor(bi, m, 32);
        if (ov > bv || (ov == bv && oi < bi)) { bv = ov; bi = oi; }
      }
      if (l == 0) sfar = bi;
    }
    __syncthreads();
    far = sfar;
    __syncthreads();
  }
}

// ---------------- kernel 2: kNN (top-16 smallest distances) ----------------
__global__ void __launch_bounds__(256) knn_kernel(const float* __restrict__ xyz,
                                                  const int* __restrict__ fps_idx,
                                                  int* __restrict__ knn_idx,
                                                  float* __restrict__ new_xyz) {
  const int b = blockIdx.y;
  const int q = blockIdx.x * 256 + threadIdx.x;
  const float* base = xyz + (size_t)b * NPTS * 3;

  __shared__ float sx[2048], sy[2048], sz[2048];

  const int fi = fps_idx[b * NS + q];
  const float qx = base[fi * 3 + 0], qy = base[fi * 3 + 1], qz = base[fi * 3 + 2];
  new_xyz[(size_t)(b * NS + q) * 3 + 0] = qx;
  new_xyz[(size_t)(b * NS + q) * 3 + 1] = qy;
  new_xyz[(size_t)(b * NS + q) * 3 + 2] = qz;

  float bd[NK];
  int   bi[NK];
#pragma unroll
  for (int j = 0; j < NK; ++j) { bd[j] = 3.0e38f; bi[j] = 0; }

  for (int chunk = 0; chunk < NPTS / 2048; ++chunk) {
    __syncthreads();
    for (int i = threadIdx.x; i < 2048; i += 256) {
      int p = chunk * 2048 + i;
      sx[i] = base[p * 3 + 0];
      sy[i] = base[p * 3 + 1];
      sz[i] = base[p * 3 + 2];
    }
    __syncthreads();
#pragma unroll 4
    for (int i = 0; i < 2048; ++i) {
      float dx = qx - sx[i], dy = qy - sy[i], dz = qz - sz[i];
      float d  = dx * dx + dy * dy + dz * dz;
      if (d < bd[NK - 1]) {
        bd[NK - 1] = d;
        bi[NK - 1] = chunk * 2048 + i;
#pragma unroll
        for (int j = NK - 1; j > 0; --j) {
          if (bd[j] < bd[j - 1]) {
            float tv = bd[j]; bd[j] = bd[j - 1]; bd[j - 1] = tv;
            int   ti = bi[j]; bi[j] = bi[j - 1]; bi[j - 1] = ti;
          }
        }
      }
    }
  }
#pragma unroll
  for (int j = 0; j < NK; ++j) knn_idx[(size_t)(b * NS + q) * NK + j] = bi[j];
}

// ---------------- kernel 3: gather + transformer block + fc + maxpool ----------------
// 1 wave == 1 group. Weights pre-swizzled into LDS in B-fragment layout (56 frags * 1KB).
// Per-wave LDS scratch: f(1024) x(1024) q(1024) k(1024) v(2048,rows16..31=0) s(512) [res aliases q]
__global__ void __launch_bounds__(128) td_main_kernel(
    const float* __restrict__ xyz, const float* __restrict__ feature,
    const float* __restrict__ w_fc1, const float* __restrict__ b_fc1,
    const float* __restrict__ w_q, const float* __restrict__ w_k, const float* __restrict__ w_v,
    const float* __restrict__ w_fc2, const float* __restrict__ b_fc2,
    const float* __restrict__ bn0_g, const float* __restrict__ bn0_b,
    const float* __restrict__ bn0_m, const float* __restrict__ bn0_v,
    const float* __restrict__ w_fc, const float* __restrict__ b_fc,
    const float* __restrict__ bn1_g, const float* __restrict__ bn1_b,
    const float* __restrict__ bn1_m, const float* __restrict__ bn1_v,
    const int* __restrict__ knn_idx, const float* __restrict__ new_xyz,
    float* __restrict__ out_feat) {
  extern __shared__ __align__(32) char smem_raw[];
  f16* smem = (f16*)smem_raw;

  const int lane   = threadIdx.x & 31;
  const int wave   = threadIdx.x >> 5;
  const int nwaves = blockDim.x >> 5;
  const int half   = lane >> 4;

  f16* wswz = smem;  // 56 fragments * 512 f16
  f16* wsc  = smem + 56 * 512 + wave * 6656;
  f16* fbuf = wsc;            // [16][64]
  f16* xbuf = wsc + 1024;     // [16][64]  (also f2)
  f16* qbuf = wsc + 2048;     // [16][64]  (also res)
  f16* kbuf = wsc + 3072;     // [16][64]
  f16* vbuf = wsc + 4096;     // [32][64], rows 16..31 zero
  f16* sbuf = wsc + 6144;     // [16][32], cols 16..31 zero
  f16* resbuf = qbuf;

  // ---- swizzle weights into fragment-native layout ----
  for (int fi = wave; fi < 56; fi += nwaves) {
    const float* W;
    int stride, koff, noff;
    if (fi < 40) {
      int m  = fi >> 3;
      int r  = fi & 7;
      int kt = r >> 2, nt = r & 3;
      if (m == 0) W = w_fc1;
      else if (m == 1) W = w_q;
      else if (m == 2) W = w_k;
      else if (m == 3) W = w_v;
      else W = w_fc2;
      stride = 64; koff = kt * 32; noff = nt * 16;
    } else {
      int r  = fi - 40;
      int kt = r >> 3, nt = r & 7;
      W = w_fc; stride = 128; koff = kt * 32; noff = nt * 16;
    }
    int n    = noff + (lane & 15);
    f16* dst = wswz + fi * 512 + lane * 16;
#pragma unroll
    for (int v = 0; v < 8; ++v) {
      int k = koff + 16 * half + 2 * v;
      dst[2 * v]     = (f16)W[k * stride + n];
      dst[2 * v + 1] = (f16)W[(k + 1) * stride + n];
    }
  }
  // zero pads (stay zero across groups: stores only touch the live regions)
  for (int t = lane; t < 1024; t += 32) vbuf[1024 + t] = (f16)0.f;
  for (int t = lane; t < 512; t += 32)
    if ((t & 31) >= 16) sbuf[t] = (f16)0.f;
  __syncthreads();

  const int gtotal = NB * NS;
  const int gstep  = gridDim.x * nwaves;
  for (int g = blockIdx.x * nwaves + wave; g < gtotal; g += gstep) {
    const int bb = g >> 11;  // g / NS

    // ---- gather f = [rel_xyz(3) | feat(61)] into fbuf[16][64] (f16) ----
    const int* kid = knn_idx + (size_t)g * NK;
    const float cx = new_xyz[g * 3 + 0], cy = new_xyz[g * 3 + 1], cz = new_xyz[g * 3 + 2];
    {
      int r  = lane >> 1;
      int c0 = (lane & 1) * 32;
      int idx = kid[r];
      const float* pp = xyz + ((size_t)bb * NPTS + idx) * 3;
      const float* pf = feature + ((size_t)bb * NPTS + idx) * DFEAT;
#pragma unroll
      for (int j = 0; j < 32; ++j) {
        int col = c0 + j;
        float val;
        if (col == 0)      val = pp[0] - cx;
        else if (col == 1) val = pp[1] - cy;
        else if (col == 2) val = pp[2] - cz;
        else               val = pf[col - 3];
        fbuf[r * 64 + col] = (f16)val;
      }
    }

    v16h Af0 = load_a_lds(fbuf, 64, lane);
    v16h Af1 = load_a_lds(fbuf + 32, 64, lane);

    // ---- x = f @ w_fc1 + b_fc1 ----
#pragma unroll
    for (int nt = 0; nt < 4; ++nt) {
      v8f acc = {};
      acc = WMMA(Af0, *(const v16h*)(wswz + (0 * 8 + 0 * 4 + nt) * 512 + lane * 16), acc);
      acc = WMMA(Af1, *(const v16h*)(wswz + (0 * 8 + 1 * 4 + nt) * 512 + lane * 16), acc);
      float bias = b_fc1[nt * 16 + (lane & 15)];
#pragma unroll
      for (int v = 0; v < 8; ++v) acc[v] += bias;
      store_d_f16(xbuf + nt * 16, 64, lane, acc);
    }
    v16h Ax0 = load_a_lds(xbuf, 64, lane);
    v16h Ax1 = load_a_lds(xbuf + 32, 64, lane);

    // ---- q, k, v ----
#pragma unroll
    for (int mtx = 0; mtx < 3; ++mtx) {
      f16* obuf = (mtx == 0) ? qbuf : ((mtx == 1) ? kbuf : vbuf);
#pragma unroll
      for (int nt = 0; nt < 4; ++nt) {
        v8f acc = {};
        acc = WMMA(Ax0, *(const v16h*)(wswz + ((1 + mtx) * 8 + nt) * 512 + lane * 16), acc);
        acc = WMMA(Ax1, *(const v16h*)(wswz + ((1 + mtx) * 8 + 4 + nt) * 512 + lane * 16), acc);
        store_d_f16(obuf + nt * 16, 64, lane, acc);
      }
    }

    // ---- logits = q @ k^T / sqrt(64), softmax over columns ----
    v16h Aq0 = load_a_lds(qbuf, 64, lane);
    v16h Aq1 = load_a_lds(qbuf + 32, 64, lane);
    v8f att  = {};
    att = WMMA(Aq0, load_bT_lds(kbuf, 64, lane), att);
    att = WMMA(Aq1, load_bT_lds(kbuf + 32, 64, lane), att);
#pragma unroll
    for (int v = 0; v < 8; ++v) {
      float x = att[v] * 0.125f;
      float r = x;
      r = fmaxf(r, __shfl_xor(r, 1, 32));
      r = fmaxf(r, __shfl_xor(r, 2, 32));
      r = fmaxf(r, __shfl_xor(r, 4, 32));
      r = fmaxf(r, __shfl_xor(r, 8, 32));
      float e = __expf(x - r);
      float s = e;
      s += __shfl_xor(s, 1, 32);
      s += __shfl_xor(s, 2, 32);
      s += __shfl_xor(s, 4, 32);
      s += __shfl_xor(s, 8, 32);
      att[v] = e / s;
    }
    store_d_f16(sbuf, 32, lane, att);
    v16h As = load_a_lds(sbuf, 32, lane);

    // ---- res = attn @ v  (K padded 16->32 against zeroed rows) ----
#pragma unroll
    for (int nt = 0; nt < 4; ++nt) {
      v8f acc = {};
      acc = WMMA(As, load_b_lds(vbuf + nt * 16, 64, lane), acc);
      store_d_f16(resbuf + nt * 16, 64, lane, acc);
    }
    v16h Ar0 = load_a_lds(resbuf, 64, lane);
    v16h Ar1 = load_a_lds(resbuf + 32, 64, lane);

    // ---- f2 = bn0(res @ w_fc2 + b_fc2 + f) ----
#pragma unroll
    for (int nt = 0; nt < 4; ++nt) {
      v8f acc = {};
      acc = WMMA(Ar0, *(const v16h*)(wswz + (4 * 8 + nt) * 512 + lane * 16), acc);
      acc = WMMA(Ar1, *(const v16h*)(wswz + (4 * 8 + 4 + nt) * 512 + lane * 16), acc);
      int col    = nt * 16 + (lane & 15);
      float bias = b_fc2[col];
      float sc   = bn0_g[col] * __frsqrt_rn(bn0_v[col] + BN_EPS);
      float sh   = bn0_b[col] - bn0_m[col] * sc;
#pragma unroll
      for (int v = 0; v < 8; ++v) {
        float fv = (float)fbuf[(v + 8 * half) * 64 + col];
        float y  = (acc[v] + bias + fv) * sc + sh;
        xbuf[(v + 8 * half) * 64 + col] = (f16)y;
      }
    }
    v16h Ag0 = load_a_lds(xbuf, 64, lane);
    v16h Ag1 = load_a_lds(xbuf + 32, 64, lane);

    // ---- h = relu(bn1(f2 @ w_fc + b_fc)); out = max over 16 rows ----
    float* outp = out_feat + (size_t)g * COUT;
#pragma unroll
    for (int nt = 0; nt < 8; ++nt) {
      v8f acc = {};
      acc = WMMA(Ag0, *(const v16h*)(wswz + (40 + nt) * 512 + lane * 16), acc);
      acc = WMMA(Ag1, *(const v16h*)(wswz + (40 + 8 + nt) * 512 + lane * 16), acc);
      int col    = nt * 16 + (lane & 15);
      float bias = b_fc[col];
      float sc   = bn1_g[col] * __frsqrt_rn(bn1_v[col] + BN_EPS);
      float sh   = bn1_b[col] - bn1_m[col] * sc;
      float mx   = -3.4e38f;
#pragma unroll
      for (int v = 0; v < 8; ++v) {
        float y = fmaxf(0.f, (acc[v] + bias) * sc + sh);
        mx = fmaxf(mx, y);
      }
      mx = fmaxf(mx, __shfl_xor(mx, 16, 32));  // lane L and L^16 share column
      if (lane < 16) outp[col] = mx;
    }
  }
}

// ---------------- host-side launch ----------------
extern "C" void kernel_launch(void* const* d_in, const int* in_sizes, int n_in,
                              void* d_out, int out_size, void* d_ws, size_t ws_size,
                              hipStream_t stream) {
  const float* xyz     = (const float*)d_in[0];
  const float* feature = (const float*)d_in[1];
  const float* w_fc1   = (const float*)d_in[2];
  const float* b_fc1   = (const float*)d_in[3];
  const float* w_q     = (const float*)d_in[4];
  const float* w_k     = (const float*)d_in[5];
  const float* w_v     = (const float*)d_in[6];
  const float* w_fc2   = (const float*)d_in[7];
  const float* b_fc2   = (const float*)d_in[8];
  const float* bn0_g   = (const float*)d_in[9];
  const float* bn0_b   = (const float*)d_in[10];
  const float* bn0_m   = (const float*)d_in[11];
  const float* bn0_v   = (const float*)d_in[12];
  const float* w_fc    = (const float*)d_in[13];
  const float* b_fc    = (const float*)d_in[14];
  const float* bn1_g   = (const float*)d_in[15];
  const float* bn1_b   = (const float*)d_in[16];
  const float* bn1_m   = (const float*)d_in[17];
  const float* bn1_v   = (const float*)d_in[18];

  float* out_xyz  = (float*)d_out;                        // [B,S,3]
  float* out_feat = out_xyz + (size_t)NB * NS * 3;        // [B,S,128]

  int* fps_idx = (int*)d_ws;                              // [B,S]
  int* knn_idx = fps_idx + NB * NS;                       // [B,S,16]

  fps_kernel<<<NB, 1024, 0, stream>>>(xyz, fps_idx);
  knn_kernel<<<dim3(NS / 256, NB), 256, 0, stream>>>(xyz, fps_idx, knn_idx, out_xyz);

  size_t smem_bytes = (size_t)(56 * 512 + 4 * 6656) * sizeof(f16);  // ~108 KB
  td_main_kernel<<<512, 128, smem_bytes, stream>>>(
      xyz, feature, w_fc1, b_fc1, w_q, w_k, w_v, w_fc2, b_fc2,
      bn0_g, bn0_b, bn0_m, bn0_v, w_fc, b_fc, bn1_g, bn1_b, bn1_m, bn1_v,
      knn_idx, out_xyz, out_feat);
}
